// GraphAttentionLayer_60344290509144
// MI455X (gfx1250) — compile-verified, matrix operand order checked by reading, not run on hardware
//
#include <hip/hip_runtime.h>
#include <hip/hip_bf16.h>
#include <hip/hip_fp16.h>

#define N_NODES 8192
#define F_DIM   128
#define KC      32              // columns per k-step (WMMA K)
#define GAT_ALPHA 0.2f

typedef _Float16 v16h __attribute__((ext_vector_type(16)));
typedef float    v8f  __attribute__((ext_vector_type(8)));
typedef unsigned int v4u __attribute__((ext_vector_type(4)));
typedef int          v8i __attribute__((ext_vector_type(8)));
typedef int          v4i __attribute__((ext_vector_type(4)));

#if __has_builtin(__builtin_amdgcn_tensor_load_to_lds)
#define USE_TDM 1
#else
#define USE_TDM 0
#endif

// ---------------- workspace layout (bytes) ----------------
// [0)          int   e2max key (monotone int encoding of float)
// [16)         float e1[8192]
// [16+32768)   float e2[8192]
// [16+65536)   _Float16 WhT[128][8192]  (transposed, f16)
#define WS_E1  16
#define WS_E2  (16 + 32768)
#define WS_WHT (16 + 65536)

// -------------- async copy helpers (fallback path) --------------
__device__ __forceinline__ void async_b128(unsigned ldsOff, unsigned gOff, unsigned long long base) {
  asm volatile("global_load_async_to_lds_b128 %0, %1, %2"
               :: "v"(ldsOff), "v"(gOff), "s"(base) : "memory");
}
__device__ __forceinline__ void async_b32(unsigned ldsOff, unsigned gOff, unsigned long long base) {
  asm volatile("global_load_async_to_lds_b32 %0, %1, %2"
               :: "v"(ldsOff), "v"(gOff), "s"(base) : "memory");
}

#if USE_TDM
// -------------- TDM 2D tile load: one instruction moves tile1 x tile0 elems --------------
// D# bitfields per CDNA5 ISA ch.8 (group0: count/lds/global/type, group1: dims/strides/tile)
// This toolchain's builtin takes 6 args: (v4u g0, v8i g1, v4i g2, v4i g3, v8i g4, i32 cpol)
__device__ __forceinline__ void tdm_load_2d(unsigned lds_addr, unsigned long long gaddr,
                                            unsigned dim0, unsigned dim1, unsigned stride0,
                                            unsigned tile0, unsigned tile1, unsigned dsz_code) {
  v4u g0;
  g0.x = 1u;                                              // count=1 (valid user descriptor)
  g0.y = lds_addr;                                        // [63:32]  lds_addr
  g0.z = (unsigned)(gaddr & 0xffffffffull);               // [95:64]  global_addr lo
  g0.w = (unsigned)((gaddr >> 32) & 0x1ffffffull) | (2u << 30); // addr[56:32] | type=2
  v8i g1;
  g1[0] = (int)(dsz_code << 16);                          // wg_mask=0, data_size
  g1[1] = (int)(dim0 << 16);                              // tensor_dim0[15:0] @bit48
  g1[2] = (int)((dim0 >> 16) | (dim1 << 16));             // dim0 hi | tensor_dim1 lo @bit80
  g1[3] = (int)((dim1 >> 16) | (tile0 << 16));            // dim1 hi | tile_dim0 @bit112
  g1[4] = (int)(tile1 & 0xffffu);                         // tile_dim1 @bit128, tile_dim2=0
  g1[5] = (int)stride0;                                   // tensor_dim0_stride[31:0] @bit160
  g1[6] = 0;                                              // stride0 hi | stride1 lo
  g1[7] = 0;
  v4i z4 = {0, 0, 0, 0};                                  // groups 2/3 unused (2D tensor)
  v8i z8 = {0, 0, 0, 0, 0, 0, 0, 0};
  __builtin_amdgcn_tensor_load_to_lds(g0, g1, z4, z4, z8, 0);
}
__device__ __forceinline__ void wait_tensorcnt_1() {
#if __has_builtin(__builtin_amdgcn_s_wait_tensorcnt)
  __builtin_amdgcn_s_wait_tensorcnt(1);
#else
  asm volatile("s_wait_tensorcnt 1" ::: "memory");
#endif
}
__device__ __forceinline__ void wait_tensorcnt_0() {
#if __has_builtin(__builtin_amdgcn_s_wait_tensorcnt)
  __builtin_amdgcn_s_wait_tensorcnt(0);
#else
  asm volatile("s_wait_tensorcnt 0" ::: "memory");
#endif
}
#endif

// ---------------- init: reset e2max key ----------------
__global__ void gat_init(int* e2key) {
  if (threadIdx.x == 0 && blockIdx.x == 0) *e2key = 0x80000000;
}

// ---------------- prep: Wh, e1, e2, WhT(f16), max(e2) ----------------
__global__ __launch_bounds__(256) void gat_prep(const float* __restrict__ h,
                                                const float* __restrict__ W,
                                                const float* __restrict__ a,
                                                float* __restrict__ e1,
                                                float* __restrict__ e2,
                                                _Float16* __restrict__ whT,
                                                int* __restrict__ e2key) {
  __shared__ float hs[16][128];
  __shared__ float whs[16][128];
  const int t  = threadIdx.x;
  const int r0 = blockIdx.x * 16;

  for (int i = t; i < 16 * 128; i += 256)
    hs[i >> 7][i & 127] = h[(size_t)(r0 + (i >> 7)) * F_DIM + (i & 127)];
  __syncthreads();

  for (int i = t; i < 16 * 128; i += 256) {
    const int r = i >> 7, c = i & 127;
    float acc = 0.f;
    #pragma unroll 8
    for (int k = 0; k < 128; ++k) acc = fmaf(hs[r][k], W[k * 128 + c], acc);
    whs[r][c] = acc;
    whT[(size_t)c * N_NODES + (r0 + r)] = (_Float16)acc;   // transposed f16 copy
  }
  __syncthreads();

  if (t < 32) {                       // t<16: e1 rows, t>=16: e2 rows
    const int r = t & 15;
    const float* av = (t < 16) ? a : (a + F_DIM);
    float acc = 0.f;
    #pragma unroll 8
    for (int c = 0; c < 128; ++c) acc = fmaf(whs[r][c], av[c], acc);
    if (t < 16) {
      e1[r0 + r] = acc;
    } else {
      e2[r0 + r] = acc;
      int b = __float_as_int(acc);
      b = (b >= 0) ? b : (b ^ 0x7fffffff);   // monotone int key
      atomicMax(e2key, b);
    }
  }
}

// ---------------- main: fused mask + softmax + (att @ Wh) via WMMA ----------------
// 128 blocks x 128 threads; 4 waves, each wave owns 16 rows; all waves share column tiles.
__global__ __launch_bounds__(128) void gat_main(const int* __restrict__ adj,
                                                const float* __restrict__ e1,
                                                const float* __restrict__ e2,
                                                const _Float16* __restrict__ whT,
                                                const int* __restrict__ e2key,
                                                float* __restrict__ out) {
  __shared__ __align__(16) int            adjS[2][64 * KC];     // 64 rows x 32 cols
  __shared__ __align__(16) unsigned short whS [2][F_DIM * KC];  // 128 n  x 32 j (f16 bits)
  __shared__ __align__(16) float          e2S [2][KC];

  const int t    = threadIdx.x;
  const int lane = t & 31;
  const int wave = t >> 5;            // 0..3
  const int half = lane >> 4;         // 0/1
  const int mloc = lane & 15;
  const int r0   = blockIdx.x * 64;
  const int rowA = r0 + wave * 16 + mloc;   // row this lane owns in A-fragment layout

  // decode global max(e2)
  int kk = *e2key;
  const float e2max = __int_as_float(kk >= 0 ? kk : (kk ^ 0x7fffffff));
  const float e1r   = e1[rowA];
  float me = e1r + e2max;
  const float mrow = (me >= 0.f) ? me : GAT_ALPHA * me;   // per-row softmax shift (upper bound)

  v8f acc[8];
  #pragma unroll
  for (int nT = 0; nT < 8; ++nT) acc[nT] = v8f{0.f, 0.f, 0.f, 0.f, 0.f, 0.f, 0.f, 0.f};
  float lsum = 0.f;

  const unsigned long long adjB = (unsigned long long)adj;
  const unsigned long long whB  = (unsigned long long)whT;
  const unsigned long long e2B  = (unsigned long long)e2;
  const unsigned adjL[2] = { (unsigned)(size_t)&adjS[0][0], (unsigned)(size_t)&adjS[1][0] };
  const unsigned whL [2] = { (unsigned)(size_t)&whS [0][0], (unsigned)(size_t)&whS [1][0] };
  const unsigned e2L [2] = { (unsigned)(size_t)&e2S [0][0], (unsigned)(size_t)&e2S [1][0] };

  // issue one k-step tile set
  auto issue_tile = [&](int b, int jb) {
#if USE_TDM
    // One TDM descriptor per wave: DMA engine moves the whole tile, EXEC ignored.
    if (wave == 0) {        // WhT tile: 128 rows x 32 f16 cols, row stride 8192
      tdm_load_2d(whL[b], whB + (unsigned long long)jb * 2ull,
                  N_NODES, F_DIM, N_NODES, KC, F_DIM, 1u);
    } else if (wave == 1) { // adj tile: 64 rows x 32 i32 cols, row stride 8192
      tdm_load_2d(adjL[b], adjB + ((unsigned long long)r0 * N_NODES + (unsigned)jb) * 4ull,
                  N_NODES, N_NODES, N_NODES, KC, 64u, 2u);
    } else if (wave == 2) { // e2 slice: 32 f32
      tdm_load_2d(e2L[b], e2B + (unsigned long long)jb * 4ull,
                  N_NODES, 1u, N_NODES, KC, 0u, 2u);
    }
#else
    // fallback: per-lane async-to-LDS (9 instructions/thread)
    const unsigned rr  = (unsigned)(t >> 1);
    const unsigned cc0 = (unsigned)((t & 1) * 16);
    const unsigned gRow = (unsigned)((r0 + (int)rr) * N_NODES + jb);
    #pragma unroll
    for (int q = 0; q < 4; ++q) {
      const unsigned d = (unsigned)(t * 16 + q * 4);
      async_b128(adjL[b] + d * 4u, (gRow + cc0 + (unsigned)(q * 4)) * 4u, adjB);
    }
    #pragma unroll
    for (int q = 0; q < 4; ++q) {
      const unsigned e = (unsigned)(t * 32 + q * 8);
      const unsigned n = e >> 5, j = e & 31;
      async_b128(whL[b] + e * 2u, (n * (unsigned)N_NODES + (unsigned)jb + j) * 2u, whB);
    }
    const unsigned c = (unsigned)(lane);
    async_b32(e2L[b] + c * 4u, ((unsigned)jb + c) * 4u, e2B);
#endif
  };

  const int KT = N_NODES / KC;   // 256 k-steps
  issue_tile(0, 0);

  for (int kt = 0; kt < KT; ++kt) {
    const int b = kt & 1;
    if (kt + 1 < KT) {
      issue_tile(1 - b, (kt + 1) * KC);
#if USE_TDM
      wait_tensorcnt_1();                                 // tile kt done; kt+1 may remain
#else
      asm volatile("s_wait_asynccnt 9" ::: "memory");
#endif
    } else {
#if USE_TDM
      wait_tensorcnt_0();
#else
      asm volatile("s_wait_asynccnt 0" ::: "memory");
#endif
    }
    __syncthreads();

    const int*            adjBuf = &adjS[b][0];
    const unsigned short* whBuf  = &whS[b][0];
    const float*          e2Buf  = &e2S[b][0];

    // ---- build A fragment (16x32 f16, A-matrix VGPR layout) + row sums ----
    union { v16h v; _Float16 hh[16]; } af;
    const int rbase = (wave * 16 + mloc) * KC;
    #pragma unroll
    for (int v = 0; v < 8; ++v) {
      #pragma unroll
      for (int s = 0; s < 2; ++s) {
        const int K = ((v < 4) ? 0 : 16) + half * 8 + (v & 3) * 2 + s;
        float e = e1r + e2Buf[K];
        e = (e >= 0.f) ? e : GAT_ALPHA * e;                  // LeakyReLU
        float p = (adjBuf[rbase + K] > 0) ? __expf(e - mrow) : 0.f;
        const _Float16 ph = (_Float16)p;
        lsum += (float)ph;                                   // consistent with f16 numerator
        af.hh[v * 2 + s] = ph;
      }
    }

    // ---- 8 WMMAs: 16x128 += P(16x32) @ WhT-tile(32x128) ----
    #pragma unroll
    for (int nT = 0; nT < 8; ++nT) {
      union { v16h v; unsigned u[8]; } bf;
      const int nn = nT * 16 + mloc;
      #pragma unroll
      for (int v = 0; v < 8; ++v) {
        const int K = ((v < 4) ? 0 : 16) + half * 8 + (v & 3) * 2;  // K even -> dword read
        bf.u[v] = *(const unsigned*)&whBuf[nn * KC + K];
      }
      acc[nT] = __builtin_amdgcn_wmma_f32_16x16x32_f16(
          false, af.v, false, bf.v, (short)0, acc[nT], false, false);
    }
    __syncthreads();
  }

  // ---- combine row sums across halves, normalize, store ----
  float tot = lsum + __shfl_xor(lsum, 16, 32);   // full sum for row mloc (both halves)
  float inv[8];
  #pragma unroll
  for (int g = 0; g < 8; ++g)
    inv[g] = 1.0f / __shfl(tot, half * 8 + g, 32);  // C-layout row = g + 8*half

  #pragma unroll
  for (int nT = 0; nT < 8; ++nT) {
    #pragma unroll
    for (int g = 0; g < 8; ++g) {
      const int row = r0 + wave * 16 + half * 8 + g;
      const int col = nT * 16 + mloc;
      out[(size_t)row * F_DIM + col] = acc[nT][g] * inv[g];
    }
  }
}

// ---------------- launch ----------------
extern "C" void kernel_launch(void* const* d_in, const int* in_sizes, int n_in,
                              void* d_out, int out_size, void* d_ws, size_t ws_size,
                              hipStream_t stream) {
  const float* h   = (const float*)d_in[0];
  const int*   adj = (const int*)  d_in[1];
  const float* W   = (const float*)d_in[2];
  const float* a   = (const float*)d_in[3];
  float* out = (float*)d_out;

  char* ws = (char*)d_ws;
  int*      e2key = (int*)ws;
  float*    e1    = (float*)(ws + WS_E1);
  float*    e2    = (float*)(ws + WS_E2);
  _Float16* whT   = (_Float16*)(ws + WS_WHT);

  gat_init<<<1, 64, 0, stream>>>(e2key);
  gat_prep<<<N_NODES / 16, 256, 0, stream>>>(h, W, a, e1, e2, whT, e2key);
  gat_main<<<N_NODES / 64, 128, 0, stream>>>(adj, e1, e2, whT, e2key, out);
}
